// MessagePassingLayer_65936337928245
// MI455X (gfx1250) — matrix operand order repaired and checked by reading, not compiled, not run on hardware
//
#include <hip/hip_runtime.h>
#include <hip/hip_bf16.h>

typedef __attribute__((ext_vector_type(16))) _Float16 v16h;
typedef __attribute__((ext_vector_type(8)))  float    v8f;
typedef __attribute__((ext_vector_type(4)))  unsigned int u32x4;
typedef __attribute__((ext_vector_type(8)))  int      i32x8;
typedef __attribute__((ext_vector_type(4)))  int      i32x4;

#define BLOCK 256

// ---------------- geometry ----------------
constexpr int Bn   = 2048;  // batch
constexpr int NJ   = 64;    // nodes
constexpr int EE   = 126;   // edges
constexpr int EP   = 128;   // padded edges
constexpr int Dd   = 128;   // node dim
constexpr int EDd  = 32;    // edge dim
// FFN = 2048, processed as 16 chunks of 128 columns

// ---------------- packed f16 weight pool (halves; 512 halves / 16x16x32 B-tile) --
constexpr int OFF_EW1  = 0;                  // (288x128)  9x8  tiles
constexpr int OFF_EW2  = OFF_EW1 + 72*512;   // (128x32)   4x2
constexpr int OFF_PROJ = OFF_EW2 + 8*512;    // (32x128)   1x8
constexpr int OFF_RW   = OFF_PROJ + 8*512;   // (256x128)  8x8
constexpr int OFF_UW   = OFF_RW  + 64*512;
constexpr int OFF_CW   = OFF_UW  + 64*512;
constexpr int OFF_F1   = OFF_CW  + 64*512;   // (128x2048) 4x128
constexpr int OFF_F2   = OFF_F1  + 512*512;  // (2048x128) 64x8
// total = OFF_F2 + 512*512 = 667,648 halves = 1,335,296 bytes of d_ws

// A-fragment K index within a 32-wide chunk (ISA 7.12.2, 16-bit A 16x32)
__device__ __forceinline__ int kidx(int lane, int h) {
    return ((h & 8) << 1) | ((lane & 16) >> 1) | (h & 7);
}

__device__ __forceinline__ v8f wmma_f16(v16h a, v16h b, v8f c) {
    return __builtin_amdgcn_wmma_f32_16x16x32_f16(
        false, a, false, b, (short)0, c, false, false);
}

// A fragment from row-major f16 LDS, base already offset to (row, kchunk)
__device__ __forceinline__ v16h frag_a(const _Float16* p, int lane) {
    v16h af;
#pragma unroll
    for (int h = 0; h < 16; ++h) af[h] = p[kidx(lane, h)];
    return af;
}

// ---- Tensor Data Mover: 1-D global->LDS copy of nelem 4-byte words (D# per
// ---- CDNA5 ISA: group0 {flags, lds_addr, gaddr, type=2},
// ---- group1 {data_size=4B, tensor_dim0=tile_dim0=stride=nelem, dim1=1}).
// ---- This toolchain exposes the 6-arg builtin (groups 2/3 + extra group, cpol).
__device__ __forceinline__ void tdm_load_1d(const void* gptr,
                                            unsigned int lds_byte_addr,
                                            unsigned int nelem_dw) {
    unsigned long long ga = (unsigned long long)gptr;
    u32x4 g0;
    g0[0] = 1u;                                              // count=1, user mode
    g0[1] = lds_byte_addr;                                   // LDS dest (bytes)
    g0[2] = (unsigned int)(ga & 0xFFFFFFFFull);              // global_addr lo
    g0[3] = (unsigned int)((ga >> 32) & 0x01FFFFFFull) | (2u << 30); // hi | type=2
    i32x8 g1;
    g1[0] = (int)(2u << 16);                                 // data_size=4B
    g1[1] = (int)((nelem_dw & 0xFFFFu) << 16);               // tensor_dim0[15:0]
    g1[2] = (int)(((nelem_dw >> 16) & 0xFFFFu) | (1u << 16));// dim0 hi | dim1=1
    g1[3] = (int)((nelem_dw & 0xFFFFu) << 16);               // tile_dim0
    g1[4] = 0;                                               // tile_dim1/2 unused
    g1[5] = (int)nelem_dw;                                   // dim0_stride lo
    g1[6] = 0;
    g1[7] = 0;
    i32x4 z4 = {};
    i32x8 z8 = {};
    __builtin_amdgcn_tensor_load_to_lds(g0, g1, z4, z4, z8, 0);
}

// ------------- weight pre-pack: f32 row-major (K x N) -> per-tile B fragments --
// B layout: lane 0-15 -> n=lane, k=h<8?h:h+8; lane 16-31 -> n=lane-16, k=h<8?8+h:16+h
__global__ __launch_bounds__(256) void pack_w16(const float* __restrict__ w,
                                                _Float16* __restrict__ dst,
                                                int NT, int Ncols) {
    const int tile = blockIdx.x;
    const int kc = tile / NT, nt = tile % NT;
    const int u = threadIdx.x;
    const int lane = u >> 3;
    const int h0 = (u & 7) * 2;
    const int col = nt * 16 + (lane & 15);
    const int kch = (lane & 16) >> 1;
#pragma unroll
    for (int j = 0; j < 2; ++j) {
        int h = h0 + j;
        int k = kc * 32 + ((h & 8) << 1) + kch + (h & 7);
        dst[(size_t)tile * 512 + lane * 16 + h] =
            (_Float16)w[(size_t)k * Ncols + col];
    }
}

// ------------- GEMM helper: (64x256)@(256x128), A=[a0|a1], double-buffered ----
__device__ __forceinline__ void mm_64x256x128(const _Float16* a0, const _Float16* a1,
                                              const _Float16* wt, int nt, int lane,
                                              v8f acc[4]) {
    const int l15 = lane & 15;
    v16h bf = *(const v16h*)(wt + (size_t)nt * 512 + lane * 16);
#pragma unroll
    for (int kc = 0; kc < 8; ++kc) {
        v16h bfn = bf;
        if (kc + 1 < 8)
            bfn = *(const v16h*)(wt + (size_t)((kc + 1) * 8 + nt) * 512 + lane * 16);
        const _Float16* abase = (kc < 4) ? (a0 + kc * 32) : (a1 + (kc - 4) * 32);
        v16h af = frag_a(abase + l15 * Dd, lane);
#pragma unroll
        for (int mt = 0; mt < 4; ++mt) {
            v16h afn = af;
            if (mt + 1 < 4) afn = frag_a(abase + ((mt + 1) * 16 + l15) * Dd, lane);
            acc[mt] = wmma_f16(af, bf, acc[mt]);
            af = afn;
        }
        bf = bfn;
    }
}

// =================== fused per-batch message-passing layer =====================
__global__ __launch_bounds__(BLOCK) void mpl_fused(
    const float* __restrict__ nf, const float* __restrict__ ea,
    const int* __restrict__ srci, const int* __restrict__ tgti,
    const float* __restrict__ em,
    const float* __restrict__ e_b1, const float* __restrict__ e_b2,
    const float* __restrict__ proj_b, const float* __restrict__ r_b,
    const float* __restrict__ u_b, const float* __restrict__ c_b,
    const float* __restrict__ n1_g, const float* __restrict__ n1_b,
    const float* __restrict__ n2_g, const float* __restrict__ n2_b,
    const float* __restrict__ f_b1, const float* __restrict__ f_b2,
    const _Float16* __restrict__ wp,
    float* __restrict__ out1, float* __restrict__ out2)
{
    // ---- LDS pool with lifetime-based aliasing (~142 KB of 320 KB WGP LDS) ----
    __shared__ __align__(16) char smem[145408];
    float*    S_NF    = (float*)(smem + 0);          // 32KB raw features (P0-P6)
    _Float16* S_NORMH = (_Float16*)(smem + 32768);   // 16KB normed f16   (P0-P6)
    _Float16* S_HC    = (_Float16*)(smem + 32768);   //      FFN chunk, reuses ^
    _Float16* S_H     = (_Float16*)(smem + 49152);   // 32KB edge hidden  (P1-P2)
    float*    S_U     = (float*)(smem + 49152);      //      update gate  (P5-P6)
    float*    S_X     = (float*)(smem + 49152);      //      x            (P6-P9)
    _Float16* S_EA    = (_Float16*)(smem + 81920);   // 8KB  edge attr    (P0-P1)
    _Float16* S_N2H   = (_Float16*)(smem + 81920);   // 16KB LN2 f16      (P7-P8)
    _Float16* S_MSG   = (_Float16*)(smem + 90112);   // 8KB  msgs f16     (P2-P3)
    float*    S_EASTG = (float*)(smem + 98304);      // 16KB TDM staging  (P0 only)
    float*    S_AGG   = (float*)(smem + 98304);      // 8KB  agg f32      (P3)
    _Float16* S_AGGH  = (_Float16*)(smem + 106496);  // 4KB  agg f16      (P3-P4)
    _Float16* S_PROJH = (_Float16*)(smem + 110592);  // 16KB proj f16     (P4-P6)
    _Float16* S_RNH   = (_Float16*)(smem + 126976);  // 16KB reset*normed (P5-P6)
    int*      S_SRC   = (int*)(smem + 143360);
    int*      S_TGT   = (int*)(smem + 143872);
    float*    S_MASK  = (float*)(smem + 144384);
    float*    S_INVC  = (float*)(smem + 144896);

    const int t    = threadIdx.x;
    const int lane = t & 31;
    const int wv   = t >> 5;           // 8 waves; wave owns one 16-col N tile
    const int b    = blockIdx.x;
    const int l15  = lane & 15;
    const int hi8  = (lane & 16) >> 1; // +8 row offset for D fragments

    // -------- P0: TDM bulk copies + indices/mask, then LN1 + degree counts -----
    if (wv == 0) {   // wave-level TDM issue (EXEC ignored, one op per wave)
        tdm_load_1d(nf + (size_t)b * NJ * Dd,
                    (unsigned int)(size_t)(void*)S_NF, NJ * Dd);
        tdm_load_1d(ea + (size_t)b * EE * EDd,
                    (unsigned int)(size_t)(void*)S_EASTG, EE * EDd);
    }
    if (t < EP) {
        S_SRC[t]  = (t < EE) ? srci[t] * Dd : 0;   // pre-scaled row offsets
        S_TGT[t]  = (t < EE) ? tgti[t] : 0;
        S_MASK[t] = (t < EE) ? em[(size_t)b * EE + t] : 0.f;
    }
    if (wv == 0) __builtin_amdgcn_s_wait_tensorcnt((short)0);
    __syncthreads();

    // edge_attr f32 staging -> f16 (pad rows 126..127 with zero)
    for (int i = t; i < EP * EDd; i += BLOCK)
        S_EA[i] = (_Float16)((i < EE * EDd) ? S_EASTG[i] : 0.f);

    {   // LN1: 4 lanes per node, __shfl_xor tree within each 4-lane group
        int node = t >> 2, part = t & 3;
        const float* row = S_NF + node * Dd;
        float s = 0.f, s2 = 0.f;
#pragma unroll
        for (int i = part * 32; i < part * 32 + 32; ++i) {
            float x = row[i]; s += x; s2 += x * x;
        }
        s += __shfl_xor(s, 1); s2 += __shfl_xor(s2, 1);
        s += __shfl_xor(s, 2); s2 += __shfl_xor(s2, 2);
        float m  = s * (1.f / Dd);
        float rs = rsqrtf(s2 * (1.f / Dd) - m * m + 1e-5f);
        for (int i = part * 32; i < part * 32 + 32; ++i)
            S_NORMH[node * Dd + i] = (_Float16)((row[i] - m) * rs * n1_g[i] + n1_b[i]);
    }
    if (t < NJ) {   // in-degree counts -> 1/clip(count,1)
        int c = 0;
        for (int e = 0; e < EE; ++e) c += (S_TGT[e] == t);
        S_INVC[t] = 1.f / fmaxf((float)c, 1.f);
    }
    __syncthreads();

    // -------- P1: edge MLP1  h = relu([src|tgt|ea] @ e_w1 + b1)  (128x288x128) --
    {
        const int nt = wv;
        int sr[8], tr[8];
#pragma unroll
        for (int mt = 0; mt < 8; ++mt) {
            int e = mt * 16 + l15;
            sr[mt] = S_SRC[e];          // already * Dd
            tr[mt] = S_TGT[e] * Dd;
        }
        v8f acc[8] = {};
        v16h bf = *(const v16h*)(wp + OFF_EW1 + (size_t)wv * 512 + lane * 16);
#pragma unroll
        for (int kc = 0; kc < 9; ++kc) {
            v16h bfn = bf;
            if (kc + 1 < 9)
                bfn = *(const v16h*)(wp + OFF_EW1 +
                        (size_t)((kc + 1) * 8 + nt) * 512 + lane * 16);
            const _Float16* p0 =
                (kc < 4) ? (S_NORMH + sr[0] + kc * 32)
              : (kc < 8) ? (S_NORMH + tr[0] + (kc - 4) * 32)
                         : (S_EA + l15 * EDd);
            v16h af = frag_a(p0, lane);
#pragma unroll
            for (int mt = 0; mt < 8; ++mt) {
                v16h afn = af;
                if (mt + 1 < 8) {
                    const _Float16* p =
                        (kc < 4) ? (S_NORMH + sr[mt + 1] + kc * 32)
                      : (kc < 8) ? (S_NORMH + tr[mt + 1] + (kc - 4) * 32)
                                 : (S_EA + ((mt + 1) * 16 + l15) * EDd);
                    afn = frag_a(p, lane);
                }
                acc[mt] = wmma_f16(af, bf, acc[mt]);
                af = afn;
            }
            bf = bfn;
        }
        float bias = e_b1[nt * 16 + l15];
#pragma unroll
        for (int mt = 0; mt < 8; ++mt)
#pragma unroll
            for (int r = 0; r < 8; ++r) {
                int m = mt * 16 + r + hi8;
                S_H[m * Dd + nt * 16 + l15] = (_Float16)fmaxf(acc[mt][r] + bias, 0.f);
            }
    }
    __syncthreads();

    // -------- P2: msgs = (h @ e_w2 + b2) * mask  (128x128x32), write out2 ------
    {
        const int mt = wv;
        v8f acc[2] = {};
        v16h af = frag_a(S_H + (mt * 16 + l15) * Dd, lane);
#pragma unroll
        for (int kc = 0; kc < 4; ++kc) {
            v16h afn = af;
            if (kc + 1 < 4)
                afn = frag_a(S_H + (mt * 16 + l15) * Dd + (kc + 1) * 32, lane);
#pragma unroll
            for (int nt = 0; nt < 2; ++nt) {
                v16h bf = *(const v16h*)(wp + OFF_EW2 +
                            (size_t)(kc * 2 + nt) * 512 + lane * 16);
                acc[nt] = wmma_f16(af, bf, acc[nt]);
            }
            af = afn;
        }
#pragma unroll
        for (int nt = 0; nt < 2; ++nt) {
            float bias = e_b2[nt * 16 + l15];
#pragma unroll
            for (int r = 0; r < 8; ++r) {
                int e = mt * 16 + r + hi8;
                int n = nt * 16 + l15;
                float v = (e < EE) ? (acc[nt][r] + bias) * S_MASK[e] : 0.f;
                S_MSG[e * EDd + n] = (_Float16)v;
                if (e < EE) out2[((size_t)b * EE + e) * EDd + n] = v;
            }
        }
    }
    __syncthreads();

    // -------- P3: scatter-mean into agg (ds_add_f32 atomics), convert f16 ------
    for (int i = t; i < NJ * EDd; i += BLOCK) S_AGG[i] = 0.f;
    __syncthreads();
    {
        int col = t & 31;
        for (int e = t >> 5; e < EE; e += 8)
            atomicAdd(&S_AGG[S_TGT[e] * EDd + col], (float)S_MSG[e * EDd + col]);
    }
    __syncthreads();
    for (int i = t; i < NJ * EDd; i += BLOCK)
        S_AGGH[i] = (_Float16)(S_AGG[i] * S_INVC[i >> 5]);
    __syncthreads();

    // -------- P4: proj = agg @ proj_w + b  (64x32x128) -------------------------
    {
        const int nt = wv;
        v16h bf = *(const v16h*)(wp + OFF_PROJ + (size_t)nt * 512 + lane * 16);
        float bias = proj_b[nt * 16 + l15];
        v16h af = frag_a(S_AGGH + l15 * EDd, lane);
#pragma unroll
        for (int mt = 0; mt < 4; ++mt) {
            v16h afn = af;
            if (mt + 1 < 4) afn = frag_a(S_AGGH + ((mt + 1) * 16 + l15) * EDd, lane);
            v8f acc = {};
            acc = wmma_f16(af, bf, acc);
            af = afn;
#pragma unroll
            for (int r = 0; r < 8; ++r) {
                int m = mt * 16 + r + hi8;
                S_PROJH[m * Dd + nt * 16 + l15] = (_Float16)(acc[r] + bias);
            }
        }
    }
    __syncthreads();

    // -------- P5: GRU reset & update gates  (2x 64x256x128) --------------------
    {
        const int nt = wv;
        v8f ar[4] = {};
        mm_64x256x128(S_NORMH, S_PROJH, wp + OFF_RW, nt, lane, ar);
        float rb = r_b[nt * 16 + l15];
#pragma unroll
        for (int mt = 0; mt < 4; ++mt)
#pragma unroll
            for (int r = 0; r < 8; ++r) {
                int m = mt * 16 + r + hi8, n = nt * 16 + l15;
                float rv = 1.f / (1.f + __expf(-(ar[mt][r] + rb)));
                S_RNH[m * Dd + n] = (_Float16)(rv * (float)S_NORMH[m * Dd + n]);
            }
        v8f au[4] = {};
        mm_64x256x128(S_NORMH, S_PROJH, wp + OFF_UW, nt, lane, au);
        float ub = u_b[nt * 16 + l15];
#pragma unroll
        for (int mt = 0; mt < 4; ++mt)
#pragma unroll
            for (int r = 0; r < 8; ++r) {
                int m = mt * 16 + r + hi8, n = nt * 16 + l15;
                S_U[m * Dd + n] = 1.f / (1.f + __expf(-(au[mt][r] + ub)));
            }
    }
    __syncthreads();

    // -------- P6: cand + residual combine -> x ---------------------------------
    {
        const int nt = wv;
        v8f ac[4] = {};
        mm_64x256x128(S_RNH, S_PROJH, wp + OFF_CW, nt, lane, ac);
        float cb = c_b[nt * 16 + l15];
#pragma unroll
        for (int mt = 0; mt < 4; ++mt)
#pragma unroll
            for (int r = 0; r < 8; ++r) {
                int m = mt * 16 + r + hi8, n = nt * 16 + l15;
                float cand = tanhf(ac[mt][r] + cb);
                float u = S_U[m * Dd + n];          // read before aliased write
                float x = S_NF[m * Dd + n] + (1.f - u) * (float)S_NORMH[m * Dd + n]
                          + u * cand;
                S_X[m * Dd + n] = x;
            }
    }
    __syncthreads();

    // -------- P7: LN2 -> f16 (4 lanes/node) ------------------------------------
    {
        int node = t >> 2, part = t & 3;
        const float* row = S_X + node * Dd;
        float s = 0.f, s2 = 0.f;
#pragma unroll
        for (int i = part * 32; i < part * 32 + 32; ++i) {
            float x = row[i]; s += x; s2 += x * x;
        }
        s += __shfl_xor(s, 1); s2 += __shfl_xor(s2, 1);
        s += __shfl_xor(s, 2); s2 += __shfl_xor(s2, 2);
        float m  = s * (1.f / Dd);
        float rs = rsqrtf(s2 * (1.f / Dd) - m * m + 1e-5f);
        for (int i = part * 32; i < part * 32 + 32; ++i)
            S_N2H[node * Dd + i] = (_Float16)((row[i] - m) * rs * n2_g[i] + n2_b[i]);
    }
    __syncthreads();

    // -------- P8: FFN in 16 chunks of 128 cols; P9: out = x + ffn --------------
    {
        const int nt = wv;
        v8f acc2[4];
        float b2v = f_b2[nt * 16 + l15];
#pragma unroll
        for (int mt = 0; mt < 4; ++mt)
#pragma unroll
            for (int r = 0; r < 8; ++r) acc2[mt][r] = b2v;

        for (int cc = 0; cc < 16; ++cc) {
            if (cc + 1 < 16)
                __builtin_prefetch((const void*)(wp + OFF_F1 +
                    (size_t)((cc + 1) * 8 + nt) * 512 + lane * 16), 0, 1);
            // FFN1 chunk: hc = gelu(n2 @ f_w1[:, cc*128 .. +128] + b1)
            v8f hc[4] = {};
            {
                v16h bf = *(const v16h*)(wp + OFF_F1 +
                            (size_t)(cc * 8 + nt) * 512 + lane * 16);
#pragma unroll
                for (int kc = 0; kc < 4; ++kc) {
                    v16h bfn = bf;
                    if (kc + 1 < 4)
                        bfn = *(const v16h*)(wp + OFF_F1 +
                            (size_t)((kc + 1) * 128 + cc * 8 + nt) * 512 + lane * 16);
                    v16h af = frag_a(S_N2H + l15 * Dd + kc * 32, lane);
#pragma unroll
                    for (int mt = 0; mt < 4; ++mt) {
                        v16h afn = af;
                        if (mt + 1 < 4)
                            afn = frag_a(S_N2H + ((mt + 1) * 16 + l15) * Dd + kc * 32,
                                         lane);
                        hc[mt] = wmma_f16(af, bf, hc[mt]);
                        af = afn;
                    }
                    bf = bfn;
                }
            }
            float b1v = f_b1[cc * 128 + nt * 16 + l15];
            __syncthreads();   // prior chunk's FFN2 reads of S_HC complete
#pragma unroll
            for (int mt = 0; mt < 4; ++mt)
#pragma unroll
                for (int r = 0; r < 8; ++r) {
                    int m = mt * 16 + r + hi8;
                    float x = hc[mt][r] + b1v;
                    x = 0.5f * x * (1.f + erff(x * 0.70710678f));   // exact gelu
                    S_HC[m * Dd + nt * 16 + l15] = (_Float16)x;
                }
            __syncthreads();
            // FFN2 partial: acc2 += hc @ f_w2[cc*128 .. +128, :]
            {
                v16h bf = *(const v16h*)(wp + OFF_F2 +
                            (size_t)((cc * 4) * 8 + nt) * 512 + lane * 16);
#pragma unroll
                for (int kc = 0; kc < 4; ++kc) {
                    v16h bfn = bf;
                    if (kc + 1 < 4)
                        bfn = *(const v16h*)(wp + OFF_F2 +
                            (size_t)((cc * 4 + kc + 1) * 8 + nt) * 512 + lane * 16);
                    v16h af = frag_a(S_HC + l15 * Dd + kc * 32, lane);
#pragma unroll
                    for (int mt = 0; mt < 4; ++mt) {
                        v16h afn = af;
                        if (mt + 1 < 4)
                            afn = frag_a(S_HC + ((mt + 1) * 16 + l15) * Dd + kc * 32,
                                         lane);
                        acc2[mt] = wmma_f16(af, bf, acc2[mt]);
                        af = afn;
                    }
                    bf = bfn;
                }
            }
        }
        // P9: out1 = x + ffn
#pragma unroll
        for (int mt = 0; mt < 4; ++mt)
#pragma unroll
            for (int r = 0; r < 8; ++r) {
                int m = mt * 16 + r + hi8, n = nt * 16 + l15;
                out1[(size_t)b * NJ * Dd + m * Dd + n] = S_X[m * Dd + n] + acc2[mt][r];
            }
    }
}

// =============================== launch ========================================
extern "C" void kernel_launch(void* const* d_in, const int* in_sizes, int n_in,
                              void* d_out, int out_size, void* d_ws, size_t ws_size,
                              hipStream_t stream) {
    const float* nf    = (const float*)d_in[0];
    const float* ea    = (const float*)d_in[1];
    const int*   srci  = (const int*)d_in[2];
    const int*   tgti  = (const int*)d_in[3];
    const float* em    = (const float*)d_in[4];
    const float* e_w1  = (const float*)d_in[6];
    const float* e_b1  = (const float*)d_in[7];
    const float* e_w2  = (const float*)d_in[8];
    const float* e_b2  = (const float*)d_in[9];
    const float* projw = (const float*)d_in[10];
    const float* projb = (const float*)d_in[11];
    const float* r_w   = (const float*)d_in[12];
    const float* r_b   = (const float*)d_in[13];
    const float* u_w   = (const float*)d_in[14];
    const float* u_b   = (const float*)d_in[15];
    const float* c_w   = (const float*)d_in[16];
    const float* c_b   = (const float*)d_in[17];
    const float* n1_g  = (const float*)d_in[18];
    const float* n1_b  = (const float*)d_in[19];
    const float* n2_g  = (const float*)d_in[20];
    const float* n2_b  = (const float*)d_in[21];
    const float* f_w1  = (const float*)d_in[22];
    const float* f_b1  = (const float*)d_in[23];
    const float* f_w2  = (const float*)d_in[24];
    const float* f_b2  = (const float*)d_in[25];

    _Float16* wp = (_Float16*)d_ws;   // 1.34 MB packed f16 weights (L2-resident)
    pack_w16<<<72,  256, 0, stream>>>(e_w1,  wp + OFF_EW1,  8,  128);
    pack_w16<<<8,   256, 0, stream>>>(e_w2,  wp + OFF_EW2,  2,  32);
    pack_w16<<<8,   256, 0, stream>>>(projw, wp + OFF_PROJ, 8,  128);
    pack_w16<<<64,  256, 0, stream>>>(r_w,   wp + OFF_RW,   8,  128);
    pack_w16<<<64,  256, 0, stream>>>(u_w,   wp + OFF_UW,   8,  128);
    pack_w16<<<64,  256, 0, stream>>>(c_w,   wp + OFF_CW,   8,  128);
    pack_w16<<<512, 256, 0, stream>>>(f_w1,  wp + OFF_F1,   128, 2048);
    pack_w16<<<512, 256, 0, stream>>>(f_w2,  wp + OFF_F2,   8,  128);

    float* out1 = (float*)d_out;
    float* out2 = out1 + (size_t)Bn * NJ * Dd;

    mpl_fused<<<Bn, BLOCK, 0, stream>>>(
        nf, ea, srci, tgti, em,
        e_b1, e_b2, projb, r_b, u_b, c_b,
        n1_g, n1_b, n2_g, n2_b, f_b1, f_b2,
        wp, out1, out2);
}
